// Decoder_21388937134808
// MI455X (gfx1250) — matrix operand order repaired and checked by reading, not compiled, not run on hardware
//
#include <hip/hip_runtime.h>
#include <hip/hip_bf16.h>

// ---------------------------------------------------------------------------
// Bahdanau-attention GRU decoder for MI455X (gfx1250).
// All GEMMs run on v_wmma_f32_16x16x32_f16 (f16 weights staged once, f32 acc).
// A-panel staged into LDS via gfx1250 async-to-LDS path when available.
// ---------------------------------------------------------------------------

#define V_  30000
#define E_  620
#define H_  1000
#define S_  50
#define B_  32
#define T_  50

typedef _Float16 v16h __attribute__((ext_vector_type(16)));
typedef float    v8f  __attribute__((ext_vector_type(8)));
typedef unsigned int u32x4 __attribute__((ext_vector_type(4)));
typedef int      v4i  __attribute__((ext_vector_type(4)));

#if defined(__gfx1250__) && __has_builtin(__builtin_amdgcn_global_load_async_to_lds_b128) && __has_builtin(__builtin_amdgcn_s_wait_asynccnt)
#define USE_ASYNC_LDS 1
#else
#define USE_ASYNC_LDS 0
#endif

#define KC 256   // K-chunk (halves) staged in LDS per iteration

// ---------------------------------------------------------------------------
// WMMA GEMM: C[m, n] = act( sum_k A16[m,k]*B16[n,k] + bias[n] )
//   A16: row-major [M, Kp]  (M multiple of 32, Kp multiple of KC, zero-padded)
//   B16: row-major [Np, Kp] (Np = gridDim.x*128, zero-padded rows beyond N)
//   C:   f32, element (m,n) at C[m*ldm + n]; stores guarded by n < N.
// Block = 256 thr = 8 waves -> 32(m) x 128(n); each wave: 2 acc (M=32) x 16 n,
// so every B fragment is loaded once and feeds two WMMAs.
// ---------------------------------------------------------------------------
__global__ __launch_bounds__(256) void gemm_f16_wmma(
    const _Float16* __restrict__ A, const _Float16* __restrict__ Bw,
    const float* __restrict__ bias, float* __restrict__ C,
    int N, int Kp, int ldm, int act)
{
    __shared__ __align__(16) _Float16 As[32][KC + 16];   // 17.4 KB, padded rows

    const int tid  = threadIdx.x;
    const int w    = tid >> 5;
    const int lane = tid & 31;
    const int lh   = lane >> 4;      // half-wave 0/1
    const int l16  = lane & 15;

    const int mBase = blockIdx.y * 32;
    const int n0    = blockIdx.x * 128 + w * 16;

    const u32x4* B4 = (const u32x4*)(Bw + (size_t)(n0 + l16) * Kp);

    v8f acc0 = {}, acc1 = {};
    for (int kc = 0; kc < Kp; kc += KC) {
        __syncthreads();
        // Cooperative stage of A[mBase..mBase+32, kc..kc+KC) into LDS:
        // 32 rows x 32 u32x4 = 1024 slots of 16B, 4 per thread.
        #pragma unroll
        for (int it = 0; it < 4; ++it) {
            const int slot = it * 256 + tid;
            const int row = slot >> 5, c4 = slot & 31;
            const _Float16* g = A + (size_t)(mBase + row) * Kp + kc + c4 * 8;
            _Float16* l = &As[row][c4 * 8];
#if USE_ASYNC_LDS
            __builtin_amdgcn_global_load_async_to_lds_b128(
                (__attribute__((address_space(1))) v4i*)g,
                (__attribute__((address_space(3))) v4i*)l, 0, 0);
#else
            *(u32x4*)l = *(const u32x4*)g;
#endif
        }
#if USE_ASYNC_LDS
        __builtin_amdgcn_s_wait_asynccnt(0);
#endif
        __syncthreads();

        // Prefetch this lane's next B chunk into cache while computing.
        if (kc + KC < Kp)
            __builtin_prefetch((const char*)B4 + (size_t)(kc + KC) * 2, 0, 0);

        const u32x4* Ar0 = (const u32x4*)&As[l16][0];
        const u32x4* Ar1 = (const u32x4*)&As[16 + l16][0];
        #pragma unroll
        for (int k0 = 0; k0 < KC; k0 += 32) {
            const int q0 = k0 >> 3;            // halves -> u32x4 units
            union { v16h v; u32x4 q[2]; } a0, a1, ub;
            // A 16-bit layout: lanes 0-15 elems 0..7 = K 0..7, 8..15 = K 16..23;
            // lanes 16-31 shifted by +8 within each K group.
            a0.q[0] = Ar0[q0 + lh];      a0.q[1] = Ar0[q0 + 2 + lh];
            a1.q[0] = Ar1[q0 + lh];      a1.q[1] = Ar1[q0 + 2 + lh];
            // B 16-bit layout: same K order as A (elems 0..7 = K 8*lh.., 8..15 = K 16+8*lh..).
            const int qb = (kc + k0) >> 3;
            ub.q[0] = B4[qb + lh];       ub.q[1] = B4[qb + 2 + lh];
            acc0 = __builtin_amdgcn_wmma_f32_16x16x32_f16(
                false, a0.v, false, ub.v, (short)0, acc0, false, false);
            acc1 = __builtin_amdgcn_wmma_f32_16x16x32_f16(
                false, a1.v, false, ub.v, (short)0, acc1, false, false);
        }
    }

    const int n = n0 + l16;
    if (n < N) {
        const float bv = bias ? bias[n] : 0.0f;
        #pragma unroll
        for (int r = 0; r < 8; ++r) {
            const int m = mBase + r + 8 * lh;   // C layout: VGPR r, half lh
            float v0 = acc0[r] + bv;
            float v1 = acc1[r] + bv;
            if (act == 1) { v0 = tanhf(v0); v1 = tanhf(v1); }
            C[(size_t)m * ldm + n]        = v0;
            C[(size_t)(m + 16) * ldm + n] = v1;
        }
    }
}

// f32 -> f16 weight staging with zero padding; dst is [Np, Kp] row-major,
// source element (n, k) = src[n*srcLd + srcOff + k] valid for n<N, k<K.
__global__ void convert_pad_f16(_Float16* __restrict__ dst,
                                const float* __restrict__ src,
                                int N, int K, int Kp, long total,
                                int srcLd, int srcOff)
{
    long i = (long)blockIdx.x * blockDim.x + threadIdx.x;
    if (i >= total) return;
    int k = (int)(i % Kp), n = (int)(i / Kp);
    float v = 0.0f;
    if (n < N && k < K) v = src[(size_t)n * srcLd + srcOff + k];
    dst[i] = (_Float16)v;
}

// cat(h_fwd, h_bwd) -> f16 [B, 2048] zero-padded (encoder_hidden is [2,B,H])
__global__ void cat16_kernel(const float* __restrict__ ehid, _Float16* __restrict__ dst)
{
    int i = blockIdx.x * blockDim.x + threadIdx.x;
    if (i >= B_ * 2048) return;
    int j = i & 2047, b = i >> 11;
    float v = 0.0f;
    if (j < H_)           v = ehid[(size_t)b * H_ + j];
    else if (j < 2 * H_)  v = ehid[(size_t)B_ * H_ + (size_t)b * H_ + (j - H_)];
    dst[i] = (_Float16)v;
}

// gi_emb[i] = bih[i] + sum_e emb[SOS=0, e] * gru_wih[i, e]   (loop-invariant)
__global__ void giemb_kernel(const float* __restrict__ emb,
                             const float* __restrict__ wih,
                             const float* __restrict__ bih,
                             float* __restrict__ giemb)
{
    int i = blockIdx.x * blockDim.x + threadIdx.x;
    if (i >= 3 * H_) return;
    float s = bih[i];
    const float* wr = wih + (size_t)i * (E_ + 2 * H_);
    for (int e = 0; e < E_; ++e) s += emb[e] * wr[e];
    giemb[i] = s;
}

// f32 h [B,H] -> f16 [B,1024] zero-padded
__global__ void pack_h16(const float* __restrict__ h, _Float16* __restrict__ h16)
{
    int i = blockIdx.x * blockDim.x + threadIdx.x;   // B_*1024
    int j = i & 1023, b = i >> 10;
    h16[i] = (_Float16)((j < H_) ? h[(size_t)b * H_ + j] : 0.0f);
}

// scores[b,s] = Va . tanh(q[b,:] + ua_keys[b,s,:]) + Va_b
__global__ __launch_bounds__(128) void scores_kernel(
    const float* __restrict__ q, const float* __restrict__ ua,
    const float* __restrict__ va, const float* __restrict__ vab,
    float* __restrict__ scores)
{
    const int bs = blockIdx.x;          // b*S + s
    const int b  = bs / S_;
    const float* qr = q + (size_t)b * H_;
    const float* ur = ua + (size_t)bs * H_;
    float p = 0.0f;
    for (int h = threadIdx.x; h < H_; h += 128) p += tanhf(qr[h] + ur[h]) * va[h];
    __shared__ float red[128];
    red[threadIdx.x] = p; __syncthreads();
    for (int o = 64; o > 0; o >>= 1) {
        if (threadIdx.x < o) red[threadIdx.x] += red[threadIdx.x + o];
        __syncthreads();
    }
    if (threadIdx.x == 0) scores[bs] = red[0] + vab[0];
}

// softmax over S, write attn weights to output, context -> f16 [B,2048] padded
__global__ __launch_bounds__(256) void attn_ctx_kernel(
    const float* __restrict__ scores, const float* __restrict__ enc,
    float* __restrict__ attn_out, _Float16* __restrict__ ctx16, int t)
{
    const int b = blockIdx.x;
    __shared__ float wsm[S_];
    if (threadIdx.x == 0) {
        float mx = -1e30f;
        for (int s = 0; s < S_; ++s) mx = fmaxf(mx, scores[b * S_ + s]);
        float sum = 0.0f;
        for (int s = 0; s < S_; ++s) { float e = expf(scores[b * S_ + s] - mx); wsm[s] = e; sum += e; }
        float inv = 1.0f / sum;
        for (int s = 0; s < S_; ++s) wsm[s] *= inv;
    }
    __syncthreads();
    if (threadIdx.x < S_)
        attn_out[((size_t)b * T_ + t) * S_ + threadIdx.x] = wsm[threadIdx.x];
    for (int k = threadIdx.x; k < 2048; k += 256) {
        float c = 0.0f;
        if (k < 2 * H_) {
            const float* er = enc + (size_t)b * S_ * 2 * H_ + k;
            for (int s = 0; s < S_; ++s) c += wsm[s] * er[(size_t)s * 2 * H_];
        }
        ctx16[b * 2048 + k] = (_Float16)c;
    }
}

// GRU gates + state update; writes h (f32) and padded h16
__global__ void hnew_kernel(const float* __restrict__ gi, const float* __restrict__ gh,
                            float* __restrict__ h, _Float16* __restrict__ h16)
{
    int i = blockIdx.x * blockDim.x + threadIdx.x;   // B_*1024
    int j = i & 1023, b = i >> 10;
    if (j >= H_) { h16[i] = (_Float16)0.0f; return; }
    const float* gib = gi + (size_t)b * 3 * H_;
    const float* ghb = gh + (size_t)b * 3 * H_;
    float r  = 1.0f / (1.0f + expf(-(gib[j] + ghb[j])));
    float z  = 1.0f / (1.0f + expf(-(gib[H_ + j] + ghb[H_ + j])));
    float nn = tanhf(gib[2 * H_ + j] + r * ghb[2 * H_ + j]);
    float hv = h[(size_t)b * H_ + j];
    float hn = (1.0f - z) * nn + z * hv;
    h[(size_t)b * H_ + j] = hn;
    h16[i] = (_Float16)hn;
}

// in-place log_softmax over one row of V_ logits
__global__ __launch_bounds__(256) void logsoftmax_kernel(float* __restrict__ logits)
{
    float* p = logits + (size_t)blockIdx.x * V_;
    __shared__ float red[256];
    float mx = -1e30f;
    for (int v = threadIdx.x; v < V_; v += 256) mx = fmaxf(mx, p[v]);
    red[threadIdx.x] = mx; __syncthreads();
    for (int o = 128; o > 0; o >>= 1) {
        if (threadIdx.x < o) red[threadIdx.x] = fmaxf(red[threadIdx.x], red[threadIdx.x + o]);
        __syncthreads();
    }
    mx = red[0]; __syncthreads();
    float sum = 0.0f;
    for (int v = threadIdx.x; v < V_; v += 256) sum += expf(p[v] - mx);
    red[threadIdx.x] = sum; __syncthreads();
    for (int o = 128; o > 0; o >>= 1) {
        if (threadIdx.x < o) red[threadIdx.x] += red[threadIdx.x + o];
        __syncthreads();
    }
    float lse = mx + logf(red[0]);
    for (int v = threadIdx.x; v < V_; v += 256) p[v] -= lse;
}

__global__ void copy_f32(const float* __restrict__ src, float* __restrict__ dst, int n)
{
    int i = blockIdx.x * blockDim.x + threadIdx.x;
    if (i < n) dst[i] = src[i];
}

// ---------------------------------------------------------------------------
extern "C" void kernel_launch(void* const* d_in, const int* in_sizes, int n_in,
                              void* d_out, int out_size, void* d_ws, size_t ws_size,
                              hipStream_t stream)
{
    const float* enc     = (const float*)d_in[0];   // [B,S,2H]
    const float* ehid    = (const float*)d_in[1];   // [2,B,H]
    const float* emb     = (const float*)d_in[2];   // [V,E]
    const float* Wa_w    = (const float*)d_in[3];
    const float* Wa_b    = (const float*)d_in[4];
    const float* Ua_w    = (const float*)d_in[5];
    const float* Ua_b    = (const float*)d_in[6];
    const float* Va_w    = (const float*)d_in[7];
    const float* Va_b    = (const float*)d_in[8];
    const float* gru_wih = (const float*)d_in[9];   // [3H, E+2H]
    const float* gru_whh = (const float*)d_in[10];  // [3H, H]
    const float* gru_bih = (const float*)d_in[11];
    const float* gru_bhh = (const float*)d_in[12];
    const float* out_w   = (const float*)d_in[13];  // [V, H]
    const float* out_b   = (const float*)d_in[14];
    const float* proj_w  = (const float*)d_in[15];  // [H, 2H]
    const float* proj_b  = (const float*)d_in[16];

    float* out = (float*)d_out;
    float* out_logp  = out;                                      // [B,T,V]
    float* out_hlast = out + (size_t)B_ * T_ * V_;               // [1,B,H]
    float* out_attn  = out_hlast + (size_t)B_ * H_;              // [B,T,S]

    // --- workspace bump allocator (256B aligned) ---
    char* wp = (char*)d_ws;
    auto alloc = [&](size_t bytes) -> void* {
        void* r = (void*)wp;
        wp += (bytes + 255) & ~(size_t)255;
        return r;
    };
    // Np padded to multiples of 128 (block tile width)
    _Float16* w16out = (_Float16*)alloc((size_t)30080 * 1024 * 2); // out_w   f16
    _Float16* wih16  = (_Float16*)alloc((size_t)3072  * 2048 * 2); // Wih ctx f16
    _Float16* whh16  = (_Float16*)alloc((size_t)3072  * 1024 * 2);
    _Float16* wa16   = (_Float16*)alloc((size_t)1024  * 1024 * 2);
    _Float16* ua16   = (_Float16*)alloc((size_t)1024  * 2048 * 2);
    _Float16* proj16 = (_Float16*)alloc((size_t)1024  * 2048 * 2);
    _Float16* enc16  = (_Float16*)alloc((size_t)1600  * 2048 * 2);
    _Float16* cat16  = (_Float16*)alloc((size_t)B_ * 2048 * 2);
    _Float16* h16    = (_Float16*)alloc((size_t)B_ * 1024 * 2);
    _Float16* ctx16  = (_Float16*)alloc((size_t)B_ * 2048 * 2);
    float* uakeys = (float*)alloc((size_t)B_ * S_ * H_ * 4);
    float* h      = (float*)alloc((size_t)B_ * H_ * 4);
    float* q      = (float*)alloc((size_t)B_ * H_ * 4);
    float* scores = (float*)alloc((size_t)B_ * S_ * 4);
    float* gi     = (float*)alloc((size_t)B_ * 3 * H_ * 4);
    float* gh     = (float*)alloc((size_t)B_ * 3 * H_ * 4);
    float* giemb  = (float*)alloc((size_t)3 * H_ * 4);

    const dim3 blk(256);
    auto cgrid = [](long total) { return dim3((unsigned)((total + 255) / 256)); };

    // --- stage f16 weights (once per launch) ---
    convert_pad_f16<<<cgrid((long)30080 * 1024), blk, 0, stream>>>(
        w16out, out_w, V_, H_, 1024, (long)30080 * 1024, H_, 0);
    convert_pad_f16<<<cgrid((long)3072 * 2048), blk, 0, stream>>>(
        wih16, gru_wih, 3 * H_, 2 * H_, 2048, (long)3072 * 2048, E_ + 2 * H_, E_);
    convert_pad_f16<<<cgrid((long)3072 * 1024), blk, 0, stream>>>(
        whh16, gru_whh, 3 * H_, H_, 1024, (long)3072 * 1024, H_, 0);
    convert_pad_f16<<<cgrid((long)1024 * 1024), blk, 0, stream>>>(
        wa16, Wa_w, H_, H_, 1024, (long)1024 * 1024, H_, 0);
    convert_pad_f16<<<cgrid((long)1024 * 2048), blk, 0, stream>>>(
        ua16, Ua_w, H_, 2 * H_, 2048, (long)1024 * 2048, 2 * H_, 0);
    convert_pad_f16<<<cgrid((long)1024 * 2048), blk, 0, stream>>>(
        proj16, proj_w, H_, 2 * H_, 2048, (long)1024 * 2048, 2 * H_, 0);
    convert_pad_f16<<<cgrid((long)1600 * 2048), blk, 0, stream>>>(
        enc16, enc, B_ * S_, 2 * H_, 2048, (long)1600 * 2048, 2 * H_, 0);
    cat16_kernel<<<cgrid((long)B_ * 2048), blk, 0, stream>>>(ehid, cat16);
    giemb_kernel<<<cgrid(3 * H_), blk, 0, stream>>>(emb, gru_wih, gru_bih, giemb);

    // --- loop-invariant precomputation ---
    // h0 = tanh(cat @ proj.T + proj_b)
    gemm_f16_wmma<<<dim3(8, 1), blk, 0, stream>>>(cat16, proj16, proj_b, h,
                                                  H_, 2048, H_, /*act=*/1);
    pack_h16<<<cgrid((long)B_ * 1024), blk, 0, stream>>>(h, h16);
    // ua_keys = enc @ Ua.T + Ua_b   (M = B*S = 1600)
    gemm_f16_wmma<<<dim3(8, 50), blk, 0, stream>>>(enc16, ua16, Ua_b, uakeys,
                                                   H_, 2048, H_, 0);

    // --- T decoder steps ---
    for (int t = 0; t < T_; ++t) {
        // q = h @ Wa.T + Wa_b
        gemm_f16_wmma<<<dim3(8, 1), blk, 0, stream>>>(h16, wa16, Wa_b, q,
                                                      H_, 1024, H_, 0);
        scores_kernel<<<dim3(B_ * S_), dim3(128), 0, stream>>>(q, uakeys, Va_w, Va_b, scores);
        attn_ctx_kernel<<<dim3(B_), blk, 0, stream>>>(scores, enc, out_attn, ctx16, t);
        // gi = ctx @ Wih_ctx.T + gi_emb ; gh = h @ Whh.T + bhh
        gemm_f16_wmma<<<dim3(24, 1), blk, 0, stream>>>(ctx16, wih16, giemb, gi,
                                                       3 * H_, 2048, 3 * H_, 0);
        gemm_f16_wmma<<<dim3(24, 1), blk, 0, stream>>>(h16, whh16, gru_bhh, gh,
                                                       3 * H_, 1024, 3 * H_, 0);
        hnew_kernel<<<cgrid((long)B_ * 1024), blk, 0, stream>>>(gi, gh, h, h16);
        // logits[:, t, :] = h_new @ out_w.T + out_b  (ldm = T*V interleaves t)
        gemm_f16_wmma<<<dim3(235, 1), blk, 0, stream>>>(h16, w16out, out_b,
                                                        out_logp + (size_t)t * V_,
                                                        V_, 1024, T_ * V_, 0);
    }

    // --- epilogue ---
    logsoftmax_kernel<<<dim3(B_ * T_), blk, 0, stream>>>(out_logp);
    copy_f32<<<cgrid((long)B_ * H_), blk, 0, stream>>>(h, out_hlast, B_ * H_);
}
